// DisplacementDecoder_33921651703961
// MI455X (gfx1250) — compile-verified, matrix operand order checked by reading, not run on hardware
//
#include <hip/hip_runtime.h>
#include <hip/hip_bf16.h>
#include <cstdint>
#include <cstddef>

// ---------------- problem constants (from reference) ----------------
#define GRIDN   512          // BOX*OS
#define BOXN    256
#define NPTSC   30000
#define NCLSC   2
#define NLATC   16
#define NLAYC   6
#define HIDC    256
#define LFREQ   10
#define BATCHC  4
#define NROWS   (BATCHC*NPTSC)   // 120000

typedef __bf16 v16bf __attribute__((ext_vector_type(16)));
typedef float  v8f   __attribute__((ext_vector_type(8)));

union ABCast { uint4 u[2]; v16bf v; };

// swizzled-weight layout: tile(kt,nt) -> 32 lanes * 16 bf16 contiguous per lane
constexpr int SWZ_IN       = 3  * 16 * 512;                 // 24576  (K=96 pad, N=256)
constexpr int SWZ_LAYER    = 8  * 16 * 512;                 // 65536  (K=256, N=256)
constexpr int SWZ_OUT1_OFF = SWZ_IN + NLAYC * SWZ_LAYER;    // 417792
constexpr int SWZ_TOTAL    = SWZ_OUT1_OFF + 8 * 1 * 512;    // 421888 (K=256, N=16 pad)

constexpr int MT = 64;     // rows per workgroup (4 waves x 16)
constexpr int AW = 264;    // LDS activation row stride (bf16), keeps 16B alignment

// =====================================================================
// Weight pre-swizzle: fp32 -> bf16, fragment-major so GEMM B-loads are
// two contiguous global_load_b128 per lane.
// =====================================================================
__global__ void swz_weights_kernel(const float* __restrict__ Win,
                                   const float* __restrict__ Wlay,
                                   const float* __restrict__ Wout1,
                                   __bf16* __restrict__ wswz) {
  int idx = blockIdx.x * blockDim.x + threadIdx.x;
  if (idx >= SWZ_TOTAL) return;
  int local, NT, layer = -1, matsel;
  if (idx < SWZ_IN)              { matsel = 0; local = idx;               NT = 16; }
  else if (idx < SWZ_OUT1_OFF)   { matsel = 1; local = idx - SWZ_IN;
                                   layer = local / SWZ_LAYER; local %= SWZ_LAYER; NT = 16; }
  else                           { matsel = 2; local = idx - SWZ_OUT1_OFF; NT = 1; }
  int i    = local & 15;
  int lane = (local >> 4) & 31;
  int tile = local >> 9;
  int nt = tile % NT, kt = tile / NT;
  int n = nt * 16 + (lane & 15);
  int k = kt * 32 + ((lane < 16) ? 0 : 8) + ((i < 8) ? i : 16 + (i - 8));
  float v = 0.0f;
  if (matsel == 0)      { if (k < 76) v = Win[k * 256 + n]; }          // (76,256) pad K->96
  else if (matsel == 1) { v = Wlay[layer * 65536 + k * 256 + n]; }     // (256,256)
  else                  { if (n < 3) v = Wout1[k * 3 + n]; }           // (256,3) pad N->16
  wswz[idx] = (__bf16)v;
}

// =====================================================================
// Fused MLP: feature gen -> W_in -> 6x (GEMM+ELU) -> head -> disp/pos/proj
// =====================================================================
__device__ __forceinline__ v16bf ldA(const __bf16* actS, int arow, int kt, bool lo) {
  int k0 = kt * 32 + (lo ? 0 : 8);
  ABCast u;
  u.u[0] = *(const uint4*)(actS + arow * AW + k0);        // K = k0..k0+7
  u.u[1] = *(const uint4*)(actS + arow * AW + k0 + 16);   // K = k0+16..k0+23
  return u.v;
}
__device__ __forceinline__ v16bf ldB(const __bf16* w, int kt, int nt, int NT, int lane) {
  const uint4* p = (const uint4*)(w + (size_t)(((kt * NT) + nt) * 32 + lane) * 16);
  ABCast u; u.u[0] = p[0]; u.u[1] = p[1];
  return u.v;
}

template <int KT>
__device__ __forceinline__ void gemm_slab(const __bf16* actS, const __bf16* w,
                                          int arow, bool lo, int lane, v8f acc[16]) {
#pragma unroll
  for (int nt = 0; nt < 16; ++nt) acc[nt] = 0.0f;
#pragma unroll
  for (int kt = 0; kt < KT; ++kt) {
    v16bf a = ldA(actS, arow, kt, lo);
#pragma unroll
    for (int nt = 0; nt < 16; ++nt) {
      v16bf b = ldB(w, kt, nt, 16, lane);
      acc[nt] = __builtin_amdgcn_wmma_f32_16x16x32_bf16(
          false, a, false, b, (short)0, acc[nt], false, false);
    }
  }
}

__device__ __forceinline__ void store_slab(__bf16* actS, v8f acc[16], int wr0,
                                           bool lo, int lane, bool doElu) {
  int nb = lane & 15;
#pragma unroll
  for (int nt = 0; nt < 16; ++nt) {
#pragma unroll
    for (int v = 0; v < 8; ++v) {
      float x = acc[nt][v];
      if (doElu) x = (x > 0.0f) ? x : (__expf(x) - 1.0f);
      int m = lo ? v : (v + 8);
      actS[(wr0 + m) * AW + nt * 16 + nb] = (__bf16)x;
    }
  }
}

__global__ __launch_bounds__(128) void mlp_kernel(
    const float* __restrict__ z, const float* __restrict__ ori,
    const float* __restrict__ mpos, const float* __restrict__ Wout2,
    const __bf16* __restrict__ wswz,
    float* __restrict__ proj, float* __restrict__ fpos, float* __restrict__ disp) {
  __shared__ __bf16 act[MT * AW];
  __shared__ float  outb[MT * 4];
  const int tid  = threadIdx.x;
  const int wave = tid >> 5, lane = tid & 31;
  const bool lo  = lane < 16;
  const int wr0  = wave * 16;
  const int arow = wr0 + (lane & 15);
  const long row0 = (long)blockIdx.x * MT;

  // ---- feature fill: [sin|cos encodings (60) | z (16) | pad (20)] ----
  for (int idx = tid; idx < MT * 96; idx += 128) {
    int r = idx / 96, c = idx - (idx / 96) * 96;
    long g = row0 + r;
    float v = 0.0f;
    if (g < (long)NROWS) {
      int b = (int)(g / NPTSC), n = (int)(g % NPTSC);
      if (c < 60) {
        int d = c / 20, cc = c % 20;
        int j = cc % 10; bool isCos = cc >= 10;
        float ang = mpos[n * 3 + d] * exp2f((float)j) * 3.14159265358979f;
        v = isCos ? __cosf(ang) : __sinf(ang);
      } else if (c < 76) {
        v = z[b * NLATC + (c - 60)];
      }
    }
    act[r * AW + c] = (__bf16)v;
  }
  __syncthreads();

  const __bf16* wIn  = wswz;
  const __bf16* wLay = wswz + SWZ_IN;
  const __bf16* wOut = wswz + SWZ_OUT1_OFF;

  v8f acc[16];
  // input layer: h = feat @ W_in  (K=96, no activation)
  gemm_slab<3>(act, wIn, arow, lo, lane, acc);
  store_slab(act, acc, wr0, lo, lane, false);
  __syncthreads();

  // hidden layers: h = elu(h @ W)
  for (int layer = 0; layer < NLAYC; ++layer) {
    if (layer < NLAYC - 1)
      __builtin_prefetch(wLay + (layer + 1) * SWZ_LAYER + tid * 512, 0, 1);
    gemm_slab<8>(act, wLay + layer * SWZ_LAYER, arow, lo, lane, acc);
    store_slab(act, acc, wr0, lo, lane, true);
    __syncthreads();
  }

  // head: t = elu(h @ W_out1pad)  (single N tile)
  v8f accO = 0.0f;
#pragma unroll
  for (int kt = 0; kt < 8; ++kt) {
    v16bf a = ldA(act, arow, kt, lo);
    v16bf b = ldB(wOut, kt, 0, 1, lane);
    accO = __builtin_amdgcn_wmma_f32_16x16x32_bf16(
        false, a, false, b, (short)0, accO, false, false);
  }
  int col = lane & 15;
  if (col < 3) {
#pragma unroll
    for (int v = 0; v < 8; ++v) {
      float x = accO[v];
      x = (x > 0.0f) ? x : (__expf(x) - 1.0f);
      int m = lo ? v : (v + 8);
      outb[(wr0 + m) * 4 + col] = x;
    }
  }
  __syncthreads();

  // per-row tail: disp = t @ W_out2, final = mpos + disp, proj = final @ Rm
  for (int r = tid; r < MT; r += 128) {
    long g = row0 + r;
    if (g >= (long)NROWS) continue;
    int b = (int)(g / NPTSC), n = (int)(g % NPTSC);
    float g0 = outb[r * 4 + 0], g1 = outb[r * 4 + 1], g2 = outb[r * 4 + 2];
    float d0 = g0 * Wout2[0] + g1 * Wout2[3] + g2 * Wout2[6];
    float d1 = g0 * Wout2[1] + g1 * Wout2[4] + g2 * Wout2[7];
    float d2 = g0 * Wout2[2] + g1 * Wout2[5] + g2 * Wout2[8];
    float p0 = mpos[n * 3 + 0] + d0;
    float p1 = mpos[n * 3 + 1] + d1;
    float p2 = mpos[n * 3 + 2] + d2;
    disp[g * 3 + 0] = d0; disp[g * 3 + 1] = d1; disp[g * 3 + 2] = d2;
    fpos[g * 3 + 0] = p0; fpos[g * 3 + 1] = p1; fpos[g * 3 + 2] = p2;
    float a  = ori[b * 3 + 0], be = ori[b * 3 + 1], c = ori[b * 3 + 2];
    float ca = cosf(a),  sa = sinf(a);
    float cb = cosf(be), sb = sinf(be);
    float cc = cosf(c),  sc = sinf(c);
    float R00 = ca * cb * cc - sa * sc, R01 = -ca * cb * sc - sa * cc;
    float R10 = sa * cb * cc + ca * sc, R11 = -sa * cb * sc + ca * cc;
    float R20 = -sb * cc,               R21 = sb * sc;
    proj[g * 2 + 0] = p0 * R00 + p1 * R10 + p2 * R20;
    proj[g * 2 + 1] = p0 * R01 + p1 * R11 + p2 * R21;
  }
}

// =====================================================================
// Splatting + FFT pipeline
// =====================================================================
__global__ void zero_kernel(float* p, int n) {
  int i = blockIdx.x * blockDim.x + threadIdx.x;
  if (i < n) p[i] = 0.0f;
}

__global__ void splat_kernel(const float* __restrict__ proj,
                             const float* __restrict__ amp,
                             const float* __restrict__ av,
                             float* __restrict__ imgs) {
  int g = blockIdx.x * blockDim.x + threadIdx.x;
  if (g >= NROWS) return;
  int b = g / NPTSC, n = g - b * NPTSC;
  float x = fminf(fmaxf((proj[g * 2 + 0] + 0.5f) * (float)(GRIDN - 1), 0.0f), 511.0f);
  float y = fminf(fmaxf((proj[g * 2 + 1] + 0.5f) * (float)(GRIDN - 1), 0.0f), 511.0f);
  float x0f = floorf(x), y0f = floorf(y);
  float fx = x - x0f, fy = y - y0f;
  int x0 = (int)x0f, y0 = (int)y0f;
  int x1 = min(x0 + 1, GRIDN - 1), y1 = min(y0 + 1, GRIDN - 1);
  float w00 = (1.f - fx) * (1.f - fy), w10 = fx * (1.f - fy);
  float w01 = (1.f - fx) * fy,         w11 = fx * fy;
  // wamp = amp * softmax(ampvar, axis=0)
  float a0 = av[n], a1 = av[NPTSC + n];
  float mx = fmaxf(a0, a1);
  float e0 = __expf(a0 - mx), e1 = __expf(a1 - mx);
  float inv = 1.0f / (e0 + e1);
  float w_c[2] = { amp[n] * e0 * inv, amp[NPTSC + n] * e1 * inv };
#pragma unroll
  for (int c = 0; c < NCLSC; ++c) {
    float* im = imgs + (size_t)(b * NCLSC + c) * GRIDN * GRIDN;
    float s = w_c[c];
    atomicAdd(im + y0 * GRIDN + x0, s * w00);
    atomicAdd(im + y0 * GRIDN + x1, s * w10);
    atomicAdd(im + y1 * GRIDN + x0, s * w01);
    atomicAdd(im + y1 * GRIDN + x1, s * w11);
  }
}

__global__ void pack_r2c_kernel(const float* __restrict__ src, float2* __restrict__ dst, int n) {
  int i = blockIdx.x * blockDim.x + threadIdx.x;
  if (i < n) dst[i] = make_float2(src[i], 0.0f);
}

// in-place batched line FFT (radix-2 DIT, line in LDS). dir=-1 fwd, +1 inv.
template <int N, int LOG2N>
__global__ void fft_lines_kernel(float2* data, int colpass, int dir, float scale) {
  __shared__ float2 buf[N];
  int line = blockIdx.x;
  int img = line / N, j = line - img * N;
  float2* base = data + (size_t)img * N * N;
  int tid = threadIdx.x;
  for (int e = tid; e < N; e += blockDim.x) {
    int r = __brev((unsigned)e) >> (32 - LOG2N);
    buf[e] = colpass ? base[(size_t)r * N + j] : base[(size_t)j * N + r];
  }
  __syncthreads();
  for (int len = 2; len <= N; len <<= 1) {
    int half = len >> 1;
    for (int k = tid; k < N / 2; k += blockDim.x) {
      int grp = k / half, pos = k - grp * half;
      int i0 = grp * len + pos, i1 = i0 + half;
      float ang = (float)dir * 6.283185307179586f * (float)pos / (float)len;
      float s, c; __sincosf(ang, &s, &c);
      float2 u = buf[i0], t = buf[i1];
      float tr = c * t.x - s * t.y, ti = c * t.y + s * t.x;
      buf[i0] = make_float2(u.x + tr, u.y + ti);
      buf[i1] = make_float2(u.x - tr, u.y - ti);
    }
    __syncthreads();
  }
  for (int e = tid; e < N; e += blockDim.x) {
    float2 v = buf[e]; v.x *= scale; v.y *= scale;
    if (colpass) base[(size_t)e * N + j] = v; else base[(size_t)j * N + e] = v;
  }
}

// filter * sum over classes + fftshift/crop/ifftshift (fused index math)
__global__ void combine_kernel(const float2* __restrict__ F,
                               const float* __restrict__ A,
                               const float* __restrict__ Bw,
                               float2* __restrict__ Fc) {
  int idx = blockIdx.x * blockDim.x + threadIdx.x;
  if (idx >= BATCHC * BOXN * BOXN) return;
  int j2 = idx & 255, j1 = (idx >> 8) & 255, b = idx >> 16;
  int s1 = (((j1 + 128) & 255) + 384) & 511;
  int s2 = (((j2 + 128) & 255) + 384) & 511;
  float f1 = (float)((s1 < 256) ? s1 : s1 - 512) / 512.0f;
  float f2 = (float)((s2 < 256) ? s2 : s2 - 512) / 512.0f;
  float R = f1 * f1 + f2 * f2;
  float2 sum = make_float2(0.0f, 0.0f);
#pragma unroll
  for (int c = 0; c < NCLSC; ++c) {
    float flt = A[c] * __expf(-Bw[c] * Bw[c] * R);
    float2 v = F[((size_t)(b * NCLSC + c) * GRIDN + s1) * GRIDN + s2];
    sum.x += flt * v.x; sum.y += flt * v.y;
  }
  Fc[idx] = sum;
}

__global__ void realize_kernel(float2* Fc, int n) {  // keep real part only
  int i = blockIdx.x * blockDim.x + threadIdx.x;
  if (i < n) Fc[i].y = 0.0f;
}

__global__ void phase_kernel(float2* Fc, const float* __restrict__ shift) {
  int idx = blockIdx.x * blockDim.x + threadIdx.x;
  if (idx >= BATCHC * BOXN * BOXN) return;
  int j2 = idx & 255, j1 = (idx >> 8) & 255, b = idx >> 16;
  float f1 = (float)((j1 < 128) ? j1 : j1 - 256) / 256.0f;
  float f2 = (float)((j2 < 128) ? j2 : j2 - 256) / 256.0f;
  float sx = shift[b * 2 + 0], sy = shift[b * 2 + 1];
  float ang = -6.283185307179586f * (sx * f2 + sy * f1);
  float s, c; __sincosf(ang, &s, &c);
  float2 v = Fc[idx];
  Fc[idx] = make_float2(c * v.x - s * v.y, c * v.y + s * v.x);
}

__global__ void write_real_kernel(const float2* __restrict__ Fc, float* __restrict__ out, int n) {
  int i = blockIdx.x * blockDim.x + threadIdx.x;
  if (i < n) out[i] = Fc[i].x;
}

// =====================================================================
extern "C" void kernel_launch(void* const* d_in, const int* in_sizes, int n_in,
                              void* d_out, int out_size, void* d_ws, size_t ws_size,
                              hipStream_t stream) {
  const float* z     = (const float*)d_in[0];
  const float* ori   = (const float*)d_in[1];
  const float* shift = (const float*)d_in[2];
  const float* mpos  = (const float*)d_in[3];
  const float* amp   = (const float*)d_in[4];
  const float* av    = (const float*)d_in[5];
  const float* Win   = (const float*)d_in[6];
  const float* Wlay  = (const float*)d_in[7];
  const float* Wout1 = (const float*)d_in[8];
  const float* Wout2 = (const float*)d_in[9];
  const float* A     = (const float*)d_in[10];
  const float* Bw    = (const float*)d_in[11];

  float* out  = (float*)d_out;                 // (4,256,256)
  float* fpos = out + BATCHC * BOXN * BOXN;    // (4,30000,3)
  float* disp = fpos + NROWS * 3;              // (4,30000,3)

  char* ws = (char*)d_ws;
  __bf16* wswz = (__bf16*)ws;                                 // 843,776 B
  float*  proj = (float*)(ws + (size_t)(1 << 20));            // 960,000 B
  float*  imgs = (float*)(ws + (size_t)(2 << 20));            // 8 MB  (4,2,512,512)
  float2* Fbuf = (float2*)(ws + (size_t)(10 << 20));          // 16 MB (complex 512^2 x8)
  float2* Fc   = (float2*)(ws + (size_t)(26 << 20));          // 2 MB  (complex 256^2 x4)

  // 1) weight pre-swizzle (fp32 -> fragment-major bf16)
  swz_weights_kernel<<<(SWZ_TOTAL + 255) / 256, 256, 0, stream>>>(Win, Wlay, Wout1, wswz);

  // 2) fused WMMA MLP + head + projection
  mlp_kernel<<<NROWS / MT, 128, 0, stream>>>(z, ori, mpos, Wout2, wswz, proj, fpos, disp);

  // 3) splat points into (4,2,512,512) images
  const int NIMG = BATCHC * NCLSC * GRIDN * GRIDN;  // 2,097,152
  zero_kernel<<<(NIMG + 255) / 256, 256, 0, stream>>>(imgs, NIMG);
  splat_kernel<<<(NROWS + 255) / 256, 256, 0, stream>>>(proj, amp, av, imgs);

  // 4) forward FFT2 512x512 (8 images)
  pack_r2c_kernel<<<(NIMG + 255) / 256, 256, 0, stream>>>(imgs, Fbuf, NIMG);
  fft_lines_kernel<512, 9><<<BATCHC * NCLSC * 512, 256, 0, stream>>>(Fbuf, 0, -1, 1.0f);
  fft_lines_kernel<512, 9><<<BATCHC * NCLSC * 512, 256, 0, stream>>>(Fbuf, 1, -1, 1.0f);

  // 5) filter-sum over classes + shift/crop/ishift -> Fc (4,256,256)
  const int NFC = BATCHC * BOXN * BOXN;  // 262,144
  combine_kernel<<<(NFC + 255) / 256, 256, 0, stream>>>(Fbuf, A, Bw, Fc);

  // 6) inverse FFT2 256 (norm 1/256 per pass, fold /OS^2 = 0.25)
  fft_lines_kernel<256, 8><<<BATCHC * 256, 128, 0, stream>>>(Fc, 0, +1, 1.0f / 256.0f);
  fft_lines_kernel<256, 8><<<BATCHC * 256, 128, 0, stream>>>(Fc, 1, +1, 0.25f / 256.0f);
  realize_kernel<<<(NFC + 255) / 256, 256, 0, stream>>>(Fc, NFC);

  // 7) forward FFT2 256, phase shift, inverse FFT2 256
  fft_lines_kernel<256, 8><<<BATCHC * 256, 128, 0, stream>>>(Fc, 0, -1, 1.0f);
  fft_lines_kernel<256, 8><<<BATCHC * 256, 128, 0, stream>>>(Fc, 1, -1, 1.0f);
  phase_kernel<<<(NFC + 255) / 256, 256, 0, stream>>>(Fc, shift);
  fft_lines_kernel<256, 8><<<BATCHC * 256, 128, 0, stream>>>(Fc, 0, +1, 1.0f / 256.0f);
  fft_lines_kernel<256, 8><<<BATCHC * 256, 128, 0, stream>>>(Fc, 1, +1, 1.0f / 256.0f);

  // 8) real part -> d_out image block
  write_real_kernel<<<(NFC + 255) / 256, 256, 0, stream>>>(Fc, out, NFC);
}
